// _SparseMLP_3393024163885
// MI455X (gfx1250) — matrix operand order repaired and checked by reading, not compiled, not run on hardware
//
#include <hip/hip_runtime.h>
#include <stdint.h>

#define H_DIM 2048
#define E_NUM 8
#define I_DIM 1024
#define T_TOK 4096

#define BM 32
#define BN 128
#define BK 64

typedef __attribute__((ext_vector_type(16))) __bf16 v16bf;
typedef __attribute__((ext_vector_type(8)))  float  v8f;

struct FragU { uint4 lo, hi; };   // 32 bytes == 16 bf16

static __device__ __forceinline__ unsigned short f2bf(float f) {
  uint32_t u = __float_as_uint(f);
  uint32_t r = (u + 0x7FFFu + ((u >> 16) & 1u)) >> 16;  // round-to-nearest-even
  return (unsigned short)r;
}

// ---------------- utility kernels ----------------
__global__ void zero_f32(float* __restrict__ p, int n) {
  int i = blockIdx.x * blockDim.x + threadIdx.x;
  int s = gridDim.x * blockDim.x;
  for (; i < n; i += s) p[i] = 0.f;
}
__global__ void zero_i32(int* __restrict__ p, int n) {
  int i = blockIdx.x * blockDim.x + threadIdx.x;
  if (i < n) p[i] = 0;
}
__global__ void cvt_bf16(const float* __restrict__ src, unsigned short* __restrict__ dst, int n) {
  int i = blockIdx.x * blockDim.x + threadIdx.x;
  int s = gridDim.x * blockDim.x;
  for (; i < n; i += s) dst[i] = f2bf(src[i]);
}

// ---- fp32 [KD][ND] -> bf16 transposed [ND][KD], 64x64 LDS tiles, per-expert z ----
#define TP 72   // padded LDS row stride (elems); 144B rows keep 16B alignment
__global__ __launch_bounds__(256) void cvt_transpose_bf16(
    const float* __restrict__ src, unsigned short* __restrict__ dst, int KD, int ND) {
  const int e  = blockIdx.z;
  const int n0 = blockIdx.x * 64;
  const int k0 = blockIdx.y * 64;
  src += (size_t)e * KD * ND;
  dst += (size_t)e * KD * ND;

  __shared__ __align__(16) unsigned short lds_t[64 * TP];

  const int tid = threadIdx.x;
  const int kl = tid >> 4;          // 0..15 (+r*16)
  const int nl = (tid & 15) * 4;    // 0..60
#pragma unroll
  for (int r = 0; r < 4; ++r) {
    int k = kl + r * 16;
    float4 v = *(const float4*)&src[(size_t)(k0 + k) * ND + n0 + nl];
    lds_t[(nl + 0) * TP + k] = f2bf(v.x);
    lds_t[(nl + 1) * TP + k] = f2bf(v.y);
    lds_t[(nl + 2) * TP + k] = f2bf(v.z);
    lds_t[(nl + 3) * TP + k] = f2bf(v.w);
  }
  __syncthreads();
  const int nr = tid >> 2;          // 0..63
  const int kq = (tid & 3) * 16;    // 0,16,32,48
  uint4 q0 = *(const uint4*)&lds_t[nr * TP + kq];
  uint4 q1 = *(const uint4*)&lds_t[nr * TP + kq + 8];
  uint4* d = (uint4*)&dst[(size_t)(n0 + nr) * KD + k0 + kq];
  d[0] = q0; d[1] = q1;
}

// ---------------- router: logits, top-2, dispatch lists (exact fp32) ----------------
__global__ void router_topk(const float* __restrict__ x, const float* __restrict__ rw,
                            int* __restrict__ counts, int* __restrict__ tok_list,
                            float* __restrict__ w_list) {
  const int lane = threadIdx.x & 31;
  const int wv   = threadIdx.x >> 5;
  const int t    = blockIdx.x * 8 + wv;
  float acc[E_NUM];
#pragma unroll
  for (int e = 0; e < E_NUM; ++e) acc[e] = 0.f;
  const float* xr = x + (size_t)t * H_DIM;
  for (int k = lane; k < H_DIM; k += 32) {
    float xv = xr[k];
#pragma unroll
    for (int e = 0; e < E_NUM; ++e) acc[e] = fmaf(xv, rw[e * H_DIM + k], acc[e]);
  }
#pragma unroll
  for (int off = 16; off > 0; off >>= 1) {
#pragma unroll
    for (int e = 0; e < E_NUM; ++e) acc[e] += __shfl_xor(acc[e], off, 32);
  }
  if (lane == 0) {
    int b0 = 0; float v0 = acc[0];
#pragma unroll
    for (int e = 1; e < E_NUM; ++e) if (acc[e] > v0) { v0 = acc[e]; b0 = e; }
    int b1 = -1; float v1 = -3.4e38f;
#pragma unroll
    for (int e = 0; e < E_NUM; ++e) if (e != b0 && acc[e] > v1) { v1 = acc[e]; b1 = e; }
    int p0 = atomicAdd(&counts[b0], 1);
    tok_list[b0 * T_TOK + p0] = t; w_list[b0 * T_TOK + p0] = v0;
    int p1 = atomicAdd(&counts[b1], 1);
    tok_list[b1 * T_TOK + p1] = t; w_list[b1 * T_TOK + p1] = v1;
  }
}

__global__ void scan_counts(const int* __restrict__ counts, int* __restrict__ offsets) {
  if (threadIdx.x == 0 && blockIdx.x == 0) {
    int s = 0;
    for (int e = 0; e < E_NUM; ++e) { offsets[e] = s; s += counts[e]; }
    offsets[E_NUM] = s;
  }
}

// ---------------- up-projection + SwiGLU (grouped / dense), K-pipelined ----------------
// B matrices are pre-transposed: BT [(E)][ND][KD] bf16
template <bool GATHER>
__global__ __launch_bounds__(256) void up_swiglu_kernel(
    const unsigned short* __restrict__ xb,      // [T, KD] bf16 row-major
    const unsigned short* __restrict__ BgT,
    const unsigned short* __restrict__ BuT,
    const float* __restrict__ gbias,            // [ND] or null
    const float* __restrict__ ubias,
    unsigned short* __restrict__ hbuf,          // gather: [T*K rows][ND]; dense: [T][ND]
    const int* __restrict__ counts,
    const int* __restrict__ tok_list,
    const float* __restrict__ w_list,
    const int* __restrict__ offsets,
    int KD, int ND) {
  const int e     = GATHER ? blockIdx.z : 0;
  const int mtile = blockIdx.y;
  const int N0    = blockIdx.x * BN;
  const int cnt   = GATHER ? counts[e] : T_TOK;
  if (mtile * BM >= cnt) return;
  const int hbase = GATHER ? offsets[e] : 0;

  __shared__ __align__(16) unsigned short lds_a[BM * BK];     // 4 KB
  __shared__ __align__(16) unsigned short lds_bg[BN * BK];    // 16 KB
  __shared__ __align__(16) unsigned short lds_bu[BN * BK];    // 16 KB
  __shared__ int   s_tok[BM];
  __shared__ float s_wt[BM];

  const int tid = threadIdx.x;
  if (tid < BM) {
    int pos = mtile * BM + tid;
    if (GATHER) {
      if (pos < cnt) { s_tok[tid] = tok_list[e * T_TOK + pos]; s_wt[tid] = w_list[e * T_TOK + pos]; }
      else           { s_tok[tid] = -1; s_wt[tid] = 0.f; }
    } else { s_tok[tid] = pos; s_wt[tid] = 1.f; }
  }
  __syncthreads();

  const unsigned short* Bg = BgT + (size_t)e * KD * ND;
  const unsigned short* Bu = BuT + (size_t)e * KD * ND;

  const int lane = tid & 31;
  const int wv   = tid >> 5;     // wave owns N sub-tile wv (16 cols)
  const int half = lane >> 4;
  const int ml   = lane & 15;

  v8f accg0 = {0.f,0.f,0.f,0.f,0.f,0.f,0.f,0.f}, accg1 = accg0;
  v8f accu0 = accg0, accu1 = accg0;

  // A coop load: 32 rows x 8 chunks of 8 bf16
  const int aj = tid >> 3;
  const int ac = (tid & 7) * 8;
  const int atok = s_tok[aj];
  const unsigned short* arow = (atok >= 0) ? (xb + (size_t)atok * KD + ac) : nullptr;
  // B coop load (transposed source): 128 n-rows x 2 k-halves of 32
  const int bn = tid & 127;
  const int bh = (tid >> 7) * 32;
  const unsigned short* bgrow = Bg + (size_t)(N0 + bn) * KD + bh;
  const unsigned short* burow = Bu + (size_t)(N0 + bn) * KD + bh;

  // register staging for the software pipeline
  uint4 ra, rg0, rg1, rg2, rg3, ru0, ru1, ru2, ru3;
  auto fetch = [&](int k0) {
    ra = uint4{0u, 0u, 0u, 0u};
    if (arow) ra = *(const uint4*)(arow + k0);
    const uint4* sg = (const uint4*)(bgrow + k0);
    rg0 = sg[0]; rg1 = sg[1]; rg2 = sg[2]; rg3 = sg[3];
    const uint4* su = (const uint4*)(burow + k0);
    ru0 = su[0]; ru1 = su[1]; ru2 = su[2]; ru3 = su[3];
  };

  fetch(0);
  for (int k0 = 0; k0 < KD; k0 += BK) {
    // commit current tile to LDS
    *(uint4*)&lds_a[aj * BK + ac] = ra;
    {
      uint4* d = (uint4*)&lds_bg[bn * BK + bh];
      d[0] = rg0; d[1] = rg1; d[2] = rg2; d[3] = rg3;
    }
    {
      uint4* d = (uint4*)&lds_bu[bn * BK + bh];
      d[0] = ru0; d[1] = ru1; d[2] = ru2; d[3] = ru3;
    }
    __syncthreads();
    // issue next tile's global loads; they complete while WMMAs run
    if (k0 + BK < KD) fetch(k0 + BK);

    const uint4* pa0 = (const uint4*)&lds_a[(ml) * BK];
    const uint4* pa1 = (const uint4*)&lds_a[(16 + ml) * BK];
    const uint4* pg  = (const uint4*)&lds_bg[(wv * 16 + ml) * BK];
    const uint4* pu  = (const uint4*)&lds_bu[(wv * 16 + ml) * BK];
#pragma unroll
    for (int kk = 0; kk < 2; ++kk) {
      const int q0 = kk * 4;
      FragU fa0, fa1, fbg, fbu;
      fa0.lo = pa0[q0 + half];     fa0.hi = pa0[q0 + 2 + half];
      fa1.lo = pa1[q0 + half];     fa1.hi = pa1[q0 + 2 + half];
      fbg.lo = pg[q0 + half];  fbg.hi = pg[q0 + 2 + half];
      fbu.lo = pu[q0 + half];  fbu.hi = pu[q0 + 2 + half];

      v16bf a0 = __builtin_bit_cast(v16bf, fa0);
      v16bf a1 = __builtin_bit_cast(v16bf, fa1);
      v16bf bg = __builtin_bit_cast(v16bf, fbg);
      v16bf bu = __builtin_bit_cast(v16bf, fbu);

      accg0 = __builtin_amdgcn_wmma_f32_16x16x32_bf16(false, a0, false, bg, (short)0, accg0, false, false);
      accg1 = __builtin_amdgcn_wmma_f32_16x16x32_bf16(false, a1, false, bg, (short)0, accg1, false, false);
      accu0 = __builtin_amdgcn_wmma_f32_16x16x32_bf16(false, a0, false, bu, (short)0, accu0, false, false);
      accu1 = __builtin_amdgcn_wmma_f32_16x16x32_bf16(false, a1, false, bu, (short)0, accu1, false, false);
    }
    __syncthreads();
  }

  const int nl = N0 + wv * 16 + ml;
  const float gb = gbias ? gbias[nl] : 0.f;
  const float ub = ubias ? ubias[nl] : 0.f;
#pragma unroll
  for (int s = 0; s < 2; ++s) {
    v8f g = s ? accg1 : accg0;
    v8f u = s ? accu1 : accu0;
#pragma unroll
    for (int r = 0; r < 8; ++r) {
      int mt  = s * 16 + half * 8 + r;
      int pos = mtile * BM + mt;
      if (pos < cnt) {
        float gv = g[r] + gb;
        float uv = u[r] + ub;
        float sg = gv / (1.f + __expf(-gv));   // silu
        hbuf[(size_t)(hbase + pos) * ND + nl] = f2bf(sg * uv * s_wt[mt]);
      }
    }
  }
}

// ---------------- down-projection + scatter-add, K-pipelined ----------------
template <bool GATHER>
__global__ __launch_bounds__(256) void down_kernel(
    const unsigned short* __restrict__ hbuf,    // [rows][KD] bf16
    const unsigned short* __restrict__ WdT,     // [(E)][ND][KD] bf16 (transposed)
    const float* __restrict__ dbias,            // [ND] or null
    float* __restrict__ out,                    // [T][ND] fp32
    const int* __restrict__ counts,
    const int* __restrict__ tok_list,
    const int* __restrict__ offsets,
    int KD, int ND) {
  const int e     = GATHER ? blockIdx.z : 0;
  const int mtile = blockIdx.y;
  const int N0    = blockIdx.x * BN;
  const int cnt   = GATHER ? counts[e] : T_TOK;
  if (mtile * BM >= cnt) return;
  const int hbase = GATHER ? offsets[e] : 0;

  __shared__ __align__(16) unsigned short lds_a[BM * BK];     // 4 KB
  __shared__ __align__(16) unsigned short lds_b[BN * BK];     // 16 KB
  __shared__ int s_tok[BM];

  const int tid = threadIdx.x;
  if (tid < BM) {
    int pos = mtile * BM + tid;
    s_tok[tid] = GATHER ? ((pos < cnt) ? tok_list[e * T_TOK + pos] : -1) : pos;
  }
  __syncthreads();

  const unsigned short* Wd = WdT + (size_t)e * KD * ND;

  const int lane = tid & 31;
  const int wv   = tid >> 5;
  const int half = lane >> 4;
  const int ml   = lane & 15;

  v8f acc0 = {0.f,0.f,0.f,0.f,0.f,0.f,0.f,0.f}, acc1 = acc0;

  const int aj = tid >> 3;
  const int ac = (tid & 7) * 8;
  const int apos = mtile * BM + aj;
  const unsigned short* arow =
      (apos < cnt) ? (hbuf + (size_t)(hbase + apos) * KD + ac) : nullptr;
  const int bn = tid & 127;
  const int bh = (tid >> 7) * 32;
  const unsigned short* brow = Wd + (size_t)(N0 + bn) * KD + bh;

  uint4 ra, rb0, rb1, rb2, rb3;
  auto fetch = [&](int k0) {
    ra = uint4{0u, 0u, 0u, 0u};
    if (arow) ra = *(const uint4*)(arow + k0);
    const uint4* s = (const uint4*)(brow + k0);
    rb0 = s[0]; rb1 = s[1]; rb2 = s[2]; rb3 = s[3];
  };

  fetch(0);
  for (int k0 = 0; k0 < KD; k0 += BK) {
    *(uint4*)&lds_a[aj * BK + ac] = ra;
    {
      uint4* d = (uint4*)&lds_b[bn * BK + bh];
      d[0] = rb0; d[1] = rb1; d[2] = rb2; d[3] = rb3;
    }
    __syncthreads();
    if (k0 + BK < KD) fetch(k0 + BK);

    const uint4* pa0 = (const uint4*)&lds_a[(ml) * BK];
    const uint4* pa1 = (const uint4*)&lds_a[(16 + ml) * BK];
    const uint4* pb  = (const uint4*)&lds_b[(wv * 16 + ml) * BK];
#pragma unroll
    for (int kk = 0; kk < 2; ++kk) {
      const int q0 = kk * 4;
      FragU fa0, fa1, fb;
      fa0.lo = pa0[q0 + half];    fa0.hi = pa0[q0 + 2 + half];
      fa1.lo = pa1[q0 + half];    fa1.hi = pa1[q0 + 2 + half];
      fb.lo  = pb[q0 + half]; fb.hi  = pb[q0 + 2 + half];

      v16bf a0 = __builtin_bit_cast(v16bf, fa0);
      v16bf a1 = __builtin_bit_cast(v16bf, fa1);
      v16bf b  = __builtin_bit_cast(v16bf, fb);

      acc0 = __builtin_amdgcn_wmma_f32_16x16x32_bf16(false, a0, false, b, (short)0, acc0, false, false);
      acc1 = __builtin_amdgcn_wmma_f32_16x16x32_bf16(false, a1, false, b, (short)0, acc1, false, false);
    }
    __syncthreads();
  }

  const int nl = N0 + wv * 16 + ml;
  const float db = dbias ? dbias[nl] : 0.f;
#pragma unroll
  for (int s = 0; s < 2; ++s) {
    v8f a = s ? acc1 : acc0;
#pragma unroll
    for (int r = 0; r < 8; ++r) {
      int mt  = s * 16 + half * 8 + r;
      int pos = mtile * BM + mt;
      if (pos < cnt) {
        int tok = s_tok[mt];
        atomicAdd(&out[(size_t)tok * ND + nl], a[r] + db);
      }
    }
  }
}

// ---------------- host launcher ----------------
extern "C" void kernel_launch(void* const* d_in, const int* in_sizes, int n_in,
                              void* d_out, int out_size, void* d_ws, size_t ws_size,
                              hipStream_t stream) {
  const float* x   = (const float*)d_in[0];
  const float* rw  = (const float*)d_in[1];
  const float* wg  = (const float*)d_in[2];
  const float* wu  = (const float*)d_in[3];
  const float* wd  = (const float*)d_in[4];
  const float* sgw = (const float*)d_in[5];
  const float* sgb = (const float*)d_in[6];
  const float* suw = (const float*)d_in[7];
  const float* sub = (const float*)d_in[8];
  const float* sdw = (const float*)d_in[9];
  const float* sdb = (const float*)d_in[10];
  float* out = (float*)d_out;

  char* ws = (char*)d_ws;
  size_t off = 0;
  auto alloc = [&](size_t bytes) -> char* {
    char* p = ws + off;
    off += (bytes + 255) & ~(size_t)255;
    return p;
  };
  const size_t nXH  = (size_t)T_TOK * H_DIM;
  const size_t nEHI = (size_t)E_NUM * H_DIM * I_DIM;
  const size_t nHI  = (size_t)H_DIM * I_DIM;

  unsigned short* xb    = (unsigned short*)alloc(nXH * 2);
  unsigned short* wgb   = (unsigned short*)alloc(nEHI * 2);   // [E][I][H] transposed
  unsigned short* wub   = (unsigned short*)alloc(nEHI * 2);   // [E][I][H] transposed
  unsigned short* wdb   = (unsigned short*)alloc(nEHI * 2);   // [E][H][I] transposed
  unsigned short* sgwb  = (unsigned short*)alloc(nHI * 2);    // [I][H]
  unsigned short* suwb  = (unsigned short*)alloc(nHI * 2);    // [I][H]
  unsigned short* sdwb  = (unsigned short*)alloc(nHI * 2);    // [H][I]
  unsigned short* hmoe  = (unsigned short*)alloc((size_t)T_TOK * 2 * I_DIM * 2);
  unsigned short* hsh   = (unsigned short*)alloc((size_t)T_TOK * I_DIM * 2);
  int*   counts   = (int*)alloc(E_NUM * sizeof(int));
  int*   offsets  = (int*)alloc((E_NUM + 1) * sizeof(int));
  int*   tok_list = (int*)alloc((size_t)E_NUM * T_TOK * sizeof(int));
  float* w_list   = (float*)alloc((size_t)E_NUM * T_TOK * sizeof(float));

  zero_f32<<<2048, 256, 0, stream>>>(out, T_TOK * H_DIM);
  zero_i32<<<1, 256, 0, stream>>>(counts, E_NUM);

  // activations: plain convert (A operand stays row-major)
  cvt_bf16<<<2048, 256, 0, stream>>>(x, xb, (int)nXH);
  // weights: convert + transpose so B operand is [N][K]
  cvt_transpose_bf16<<<dim3(I_DIM / 64, H_DIM / 64, E_NUM), 256, 0, stream>>>(wg,  wgb,  H_DIM, I_DIM);
  cvt_transpose_bf16<<<dim3(I_DIM / 64, H_DIM / 64, E_NUM), 256, 0, stream>>>(wu,  wub,  H_DIM, I_DIM);
  cvt_transpose_bf16<<<dim3(H_DIM / 64, I_DIM / 64, E_NUM), 256, 0, stream>>>(wd,  wdb,  I_DIM, H_DIM);
  cvt_transpose_bf16<<<dim3(I_DIM / 64, H_DIM / 64, 1),     256, 0, stream>>>(sgw, sgwb, H_DIM, I_DIM);
  cvt_transpose_bf16<<<dim3(I_DIM / 64, H_DIM / 64, 1),     256, 0, stream>>>(suw, suwb, H_DIM, I_DIM);
  cvt_transpose_bf16<<<dim3(H_DIM / 64, I_DIM / 64, 1),     256, 0, stream>>>(sdw, sdwb, I_DIM, H_DIM);

  router_topk<<<T_TOK / 8, 256, 0, stream>>>(x, rw, counts, tok_list, w_list);
  scan_counts<<<1, 1, 0, stream>>>(counts, offsets);

  // MoE experts: up+SwiGLU (combine weight folded), then down+scatter
  up_swiglu_kernel<true><<<dim3(I_DIM / BN, T_TOK / BM, E_NUM), 256, 0, stream>>>(
      xb, wgb, wub, nullptr, nullptr, hmoe, counts, tok_list, w_list, offsets, H_DIM, I_DIM);
  down_kernel<true><<<dim3(H_DIM / BN, T_TOK / BM, E_NUM), 256, 0, stream>>>(
      hmoe, wdb, nullptr, out, counts, tok_list, offsets, I_DIM, H_DIM);

  // Shared expert (identity routing, biases)
  up_swiglu_kernel<false><<<dim3(I_DIM / BN, T_TOK / BM, 1), 256, 0, stream>>>(
      xb, sgwb, suwb, sgb, sub, hsh, nullptr, nullptr, nullptr, nullptr, H_DIM, I_DIM);
  down_kernel<false><<<dim3(H_DIM / BN, T_TOK / BM, 1), 256, 0, stream>>>(
      hsh, sdwb, sdb, out, nullptr, nullptr, nullptr, I_DIM, H_DIM);

  (void)in_sizes; (void)n_in; (void)out_size; (void)ws_size;
}
